// Attention_19550691132015
// MI455X (gfx1250) — matrix operand order repaired and checked by reading, not compiled, not run on hardware
//
#include <hip/hip_runtime.h>

typedef __bf16 bf16;
typedef __attribute__((ext_vector_type(16))) __bf16 v16bf;
typedef __attribute__((ext_vector_type(8)))  __bf16 v8bf;
typedef __attribute__((ext_vector_type(4)))  __bf16 v4bf;
typedef __attribute__((ext_vector_type(8)))  float  v8f;
typedef __attribute__((ext_vector_type(4)))  unsigned int u32x4;
typedef __attribute__((ext_vector_type(8)))  int i32x8;
typedef __attribute__((ext_vector_type(4)))  int i32x4;

#define SEQ     2048
#define DIM     2048
#define BATCH   2
#define NH      32
#define NKV     8
#define HD      64

#if __has_builtin(__builtin_amdgcn_tensor_load_to_lds)
#define HAVE_TDM 1
#else
#define HAVE_TDM 0
#endif

static __device__ inline v16bf cat8(v8bf lo, v8bf hi) {
  return __builtin_shufflevector(lo, hi, 0,1,2,3,4,5,6,7,8,9,10,11,12,13,14,15);
}

static __device__ inline v8f wmma_bf16(v16bf a, v16bf b, v8f c) {
  // D = A(16x32 bf16) * B(32x16 bf16) + C(16x16 f32)
  return __builtin_amdgcn_wmma_f32_16x16x32_bf16(false, a, false, b,
                                                 (short)0, c, false, false);
}

static __device__ inline v8f vzero8() {
  v8f z;
#pragma unroll
  for (int i = 0; i < 8; ++i) z[i] = 0.0f;
  return z;
}

// Async DMA: global -> LDS, 16 bytes per lane. LDS offset is the low 32 bits
// of the generic shared-memory address.
static __device__ inline void async_copy_b128(bf16* lds_ptr, const bf16* gptr) {
  unsigned int lds_off = (unsigned int)(unsigned long long)lds_ptr;
  asm volatile("global_load_async_to_lds_b128 %0, %1, off"
               :: "v"(lds_off), "v"(gptr) : "memory");
}

static __device__ inline void wait_async_le2() {
  asm volatile("s_wait_asynccnt 0x2" ::: "memory");
}
static __device__ inline void wait_async_le1() {
  asm volatile("s_wait_asynccnt 0x1" ::: "memory");
}

#if HAVE_TDM
// One-shot TDM 2D tile copy: tile_d0 elements (contiguous) x tile_d1 lines,
// 2-byte elements, line stride = stride_elems. D# per ISA 08_async_tensor 8.3/8.4.
static __device__ inline void tdm_load_2d(bf16* lds, const bf16* g,
                                          int tile_d0, int tile_d1,
                                          int stride_elems) {
  unsigned long long ga = (unsigned long long)g;
  u32x4 g0;
  g0[0] = 1u;                                        // count=1, user mode
  g0[1] = (unsigned int)(unsigned long long)lds;     // lds_addr
  g0[2] = (unsigned int)ga;                          // global_addr[31:0]
  g0[3] = ((unsigned int)(ga >> 32) & 0x01FFFFFFu) | (2u << 30); // addr[56:32], type=2
  i32x8 g1;
  g1[0] = 0x00010000;                       // workgroup_mask=0, data_size=1 (2B)
  g1[1] = (tile_d0 & 0xffff) << 16;         // tensor_dim0[15:0] @ bits63:48
  g1[2] = (tile_d1 & 0xffff) << 16;         // tensor_dim1[15:0] @ bits95:80
  g1[3] = (tile_d0 & 0xffff) << 16;         // tile_dim0 @ bits127:112
  g1[4] = (tile_d1 & 0xffff);               // tile_dim1 @ bits143:128
  g1[5] = stride_elems;                     // tensor_dim0_stride[31:0]
  g1[6] = 0;
  g1[7] = 0;
  i32x4 z4 = {0, 0, 0, 0};
#if __clang_major__ >= 23
  i32x8 z8 = {0, 0, 0, 0, 0, 0, 0, 0};
  __builtin_amdgcn_tensor_load_to_lds(g0, g1, z4, z4, z8, 0);
#else
  __builtin_amdgcn_tensor_load_to_lds(g0, g1, z4, z4, 0);
#endif
}
#endif

// ---------------------------------------------------------------- fp32 -> bf16
__global__ void cvt_f32_bf16(const float* __restrict__ src,
                             bf16* __restrict__ dst, int n4) {
  int i = blockIdx.x * blockDim.x + threadIdx.x;
  int stride = gridDim.x * blockDim.x;
  for (; i < n4; i += stride) {
    float4 f = ((const float4*)src)[i];
    v4bf o;
    o[0] = (bf16)f.x; o[1] = (bf16)f.y; o[2] = (bf16)f.z; o[3] = (bf16)f.w;
    *(v4bf*)(dst + 4 * (size_t)i) = o;
  }
}

// ---------------------------------------------------------------- bf16 GEMM
// C[M,N] (f32) = A[M,K] (bf16, row-major) * B[K,N] (bf16, row-major)
// Block tile 128x128, BK=32, double-buffered LDS, pipelined staging.
// A tile via TDM (one 2D descriptor per tile) or per-lane async DMA fallback;
// B tile staged transposed through VGPRs.
__global__ __launch_bounds__(256)
void gemm_bf16(const bf16* __restrict__ A, const bf16* __restrict__ B,
               float* __restrict__ C, int M, int N, int K) {
  __shared__ __align__(32) bf16 As[2][128 * 32];   // [m][k]
  __shared__ __align__(32) bf16 BsT[2][128 * 32];  // [n][k]

  const int t    = threadIdx.x;
  const int lane = t & 31;
  const int wave = t >> 5;
  const int half = lane >> 4;
  const int l16  = lane & 15;
  const int wm   = wave & 3;
  const int wn   = wave >> 2;
  const int bm0  = blockIdx.y * 128;
  const int bn0  = blockIdx.x * 128;

  // B staging coordinates (2 chunks of 8 per thread)
  const int kr0 = (t * 2) >> 4, n80 = ((t * 2) & 15) * 8;
  const int kr1 = (t * 2 + 1) >> 4, n81 = ((t * 2 + 1) & 15) * 8;
  // A async fallback coordinates
  const int ar0 = (t * 2) >> 2, ak0 = ((t * 2) & 3) * 8;
  const int ar1 = (t * 2 + 1) >> 2, ak1 = ((t * 2 + 1) & 3) * 8;

  v8f acc[2][4];
#pragma unroll
  for (int s = 0; s < 2; ++s)
#pragma unroll
    for (int c = 0; c < 4; ++c) acc[s][c] = vzero8();

  v8bf breg0, breg1;
  auto issueA = [&](int k0, int buf) {
#if HAVE_TDM
    if (wave == 0)
      tdm_load_2d(&As[buf][0], A + (size_t)bm0 * K + k0, 32, 128, K);
#else
    async_copy_b128(&As[buf][ar0 * 32 + ak0], A + (size_t)(bm0 + ar0) * K + k0 + ak0);
    async_copy_b128(&As[buf][ar1 * 32 + ak1], A + (size_t)(bm0 + ar1) * K + k0 + ak1);
#endif
  };
  auto loadB = [&](int k0) {
    breg0 = *(const v8bf*)(B + (size_t)(k0 + kr0) * N + bn0 + n80);
    breg1 = *(const v8bf*)(B + (size_t)(k0 + kr1) * N + bn0 + n81);
  };
  auto storeB = [&](int buf) {
#pragma unroll
    for (int j = 0; j < 8; ++j) BsT[buf][(n80 + j) * 32 + kr0] = breg0[j];
#pragma unroll
    for (int j = 0; j < 8; ++j) BsT[buf][(n81 + j) * 32 + kr1] = breg1[j];
  };

  issueA(0, 0);
  loadB(0);

  const int nk = K / 32;
  for (int ik = 0; ik < nk; ++ik) {
    const int cur = ik & 1;
    const int knext = (ik + 1 < nk) ? (ik + 1) * 32 : 0;  // wrapped tail prefetch
    __syncthreads();                 // all waves done computing ik-1
    storeB(cur);
    issueA(knext, cur ^ 1);
    loadB(knext);
#if HAVE_TDM
    if (wave == 0) __builtin_amdgcn_s_wait_tensorcnt(1);
#else
    wait_async_le2();                // current tile's copies landed
#endif
    __syncthreads();                 // tile ik fully staged for all waves

    v16bf afr[2];
#pragma unroll
    for (int s = 0; s < 2; ++s) {
      const bf16* ap = &As[cur][(wm * 32 + s * 16 + l16) * 32 + half * 16];
      afr[s] = cat8(*(const v8bf*)ap, *(const v8bf*)(ap + 8));
    }
#pragma unroll
    for (int c = 0; c < 4; ++c) {
      const bf16* bp = &BsT[cur][(wn * 64 + c * 16 + l16) * 32 + half * 16];
      v16bf bfrag = *(const v16bf*)bp;
#pragma unroll
      for (int s = 0; s < 2; ++s)
        acc[s][c] = wmma_bf16(afr[s], bfrag, acc[s][c]);
    }
  }

  // C layout: VGPR v -> row m = v + 8*half, col n = l16
#pragma unroll
  for (int s = 0; s < 2; ++s)
#pragma unroll
    for (int c = 0; c < 4; ++c) {
      float* cp = C + (size_t)(bm0 + wm * 32 + s * 16 + half * 8) * N
                    + bn0 + wn * 64 + c * 16 + l16;
#pragma unroll
      for (int v = 0; v < 8; ++v) cp[(size_t)v * N] = acc[s][c][v];
    }
}

// ---------------------------------------------------------------- RoPE + relayout
// src: [B*S, n_heads*64] f32   ->   dst: [B, n_heads, S, 64] bf16 (rotated pairs)
__global__ void rope_scatter(const float* __restrict__ src,
                             bf16* __restrict__ dst, int n_heads) {
  int id = blockIdx.x * blockDim.x + threadIdx.x;
  int i  = id & 31;            // pair index 0..31
  int tmp = id >> 5;
  int h  = tmp % n_heads;
  tmp /= n_heads;
  int s  = tmp & (SEQ - 1);
  int b  = tmp >> 11;
  if (b >= BATCH) return;

  float inv_freq = __powf(10000.0f, -(float)i / 32.0f);
  float ang = (float)s * inv_freq;
  float c, sn;
  __sincosf(ang, &c, &sn);

  size_t srow = ((size_t)b * SEQ + s) * ((size_t)n_heads * 64) + h * 64 + 2 * i;
  float x0 = src[srow];
  float x1 = src[srow + 1];

  size_t drow = (((size_t)b * n_heads + h) * SEQ + s) * 64 + 2 * i;
  dst[drow]     = (bf16)(x0 * c - x1 * sn);
  dst[drow + 1] = (bf16)(x0 * sn + x1 * c);
}

// src: [B*S, 512] f32 -> dst: [B, 8, S, 64] bf16
__global__ void v_scatter(const float* __restrict__ src, bf16* __restrict__ dst) {
  int id = blockIdx.x * blockDim.x + threadIdx.x;
  int d  = id & 63;
  int kh = (id >> 6) & 7;
  int s  = (id >> 9) & (SEQ - 1);
  int b  = id >> 20;
  if (b >= BATCH) return;
  dst[(((size_t)b * NKV + kh) * SEQ + s) * 64 + d] =
      (bf16)src[((size_t)b * SEQ + s) * 512 + kh * 64 + d];
}

// ---------------------------------------------------------------- flash attention
// Q: [B, 32, S, 64] bf16 ; K,V: [B, 8, S, 64] bf16 ; Out: [B*S, 2048] bf16
// grid = (S/128, 32, B), 256 threads. Each wave owns 16 q rows; kv tiles of 32.
// K tile via per-lane async DMA, V transposed stage, both double-buffered.
__global__ __launch_bounds__(256)
void flash_attn(const bf16* __restrict__ Q, const bf16* __restrict__ Kb,
                const bf16* __restrict__ Vb, bf16* __restrict__ Out) {
  __shared__ __align__(32) bf16 Kt[2][32 * 64];    // [kv][d]
  __shared__ __align__(32) bf16 Vt[2][64 * 32];    // [d][kv]
  __shared__ __align__(32) bf16 Pl[8 * 16 * 32];   // per-wave P slices

  const int t    = threadIdx.x;
  const int lane = t & 31;
  const int wave = t >> 5;
  const int half = lane >> 4;
  const int l16  = lane & 15;
  const int qblk = blockIdx.x;
  const int h    = blockIdx.y;
  const int b    = blockIdx.z;
  const int kh   = h >> 2;  // 4 query heads per kv head

  const int q0 = qblk * 128 + wave * 16;
  const bf16* qbase = Q  + (((size_t)b * NH  + h)  * SEQ) * 64;
  const bf16* kbase = Kb + (((size_t)b * NKV + kh) * SEQ) * 64;
  const bf16* vbase = Vb + (((size_t)b * NKV + kh) * SEQ) * 64;

  // Q A-fragments (two k-steps over head dim)
  const bf16* qp = qbase + (size_t)(q0 + l16) * 64 + half * 16;
  v16bf qf0 = cat8(*(const v8bf*)(qp),      *(const v8bf*)(qp + 8));
  v16bf qf1 = cat8(*(const v8bf*)(qp + 32), *(const v8bf*)(qp + 40));

  v8f o[4];
#pragma unroll
  for (int c = 0; c < 4; ++c) o[c] = vzero8();
  float mrow[8], lrow[8];
#pragma unroll
  for (int v = 0; v < 8; ++v) { mrow[v] = -1e30f; lrow[v] = 0.0f; }

  bf16* Pw = Pl + wave * (16 * 32);
  const int srow = t >> 3, sd8 = (t & 7) * 8;  // staging coords
  v8bf vreg;

  auto issueK = [&](int kv0, int buf) {
    async_copy_b128(&Kt[buf][srow * 64 + sd8],
                    kbase + (size_t)(kv0 + srow) * 64 + sd8);
  };
  auto loadV = [&](int kv0) {
    vreg = *(const v8bf*)(vbase + (size_t)(kv0 + srow) * 64 + sd8);
  };
  auto storeV = [&](int buf) {
#pragma unroll
    for (int j = 0; j < 8; ++j) Vt[buf][(sd8 + j) * 32 + srow] = vreg[j];
  };

  const int ntile = (qblk * 128 + 128) / 32;
  issueK(0, 0);
  loadV(0);

  for (int it = 0; it < ntile; ++it) {
    const int cur = it & 1;
    const int kvn = (it + 1 < ntile) ? (it + 1) * 32 : 0;  // wrapped tail
    const int kv0 = it * 32;
    __syncthreads();
    storeV(cur);
    issueK(kvn, cur ^ 1);
    loadV(kvn);
    wait_async_le1();
    __syncthreads();

    // scores S[16 x 32] = Q * K^T : K rows ARE the B-fragment layout
    v8f s0 = vzero8(), s1 = vzero8();
#pragma unroll
    for (int ks = 0; ks < 2; ++ks) {
      v16bf qa = ks ? qf1 : qf0;
      const bf16* kp0 = &Kt[cur][(l16)      * 64 + ks * 32 + half * 16];
      const bf16* kp1 = &Kt[cur][(16 + l16) * 64 + ks * 32 + half * 16];
      s0 = wmma_bf16(qa, *(const v16bf*)kp0, s0);
      s1 = wmma_bf16(qa, *(const v16bf*)kp1, s1);
    }

    // scale + causal mask + online softmax (row = v + 8*half, col = l16)
#pragma unroll
    for (int v = 0; v < 8; ++v) {
      int qg = q0 + half * 8 + v;
      float a0 = s0[v] * 0.125f;
      float a1 = s1[v] * 0.125f;
      if (kv0 + l16 > qg)      a0 = -1e30f;
      if (kv0 + 16 + l16 > qg) a1 = -1e30f;

      float mx = fmaxf(a0, a1);
      mx = fmaxf(mx, __shfl_xor(mx, 1, 32));
      mx = fmaxf(mx, __shfl_xor(mx, 2, 32));
      mx = fmaxf(mx, __shfl_xor(mx, 4, 32));
      mx = fmaxf(mx, __shfl_xor(mx, 8, 32));
      float mnew = fmaxf(mrow[v], mx);
      float corr = __expf(mrow[v] - mnew);
      mrow[v] = mnew;

      float p0 = __expf(a0 - mnew);
      float p1 = __expf(a1 - mnew);
      float rs = p0 + p1;
      rs += __shfl_xor(rs, 1, 32);
      rs += __shfl_xor(rs, 2, 32);
      rs += __shfl_xor(rs, 4, 32);
      rs += __shfl_xor(rs, 8, 32);
      lrow[v] = lrow[v] * corr + rs;

      int m = half * 8 + v;
      Pw[m * 32 + l16]      = (bf16)p0;
      Pw[m * 32 + 16 + l16] = (bf16)p1;

#pragma unroll
      for (int c = 0; c < 4; ++c) o[c][v] *= corr;
    }

    // P as A-fragment (wave-local LDS round trip for the layout transpose)
    const bf16* pp = Pw + l16 * 32 + half * 16;
    v16bf pf = cat8(*(const v8bf*)pp, *(const v8bf*)(pp + 8));

    // O += P(16x32) * V(32x64)
#pragma unroll
    for (int c = 0; c < 4; ++c) {
      v16bf vf = *(const v16bf*)(&Vt[cur][(c * 16 + l16) * 32 + half * 16]);
      o[c] = wmma_bf16(pf, vf, o[c]);
    }
  }

  // normalize + write to [B*S, 2048] bf16
#pragma unroll
  for (int v = 0; v < 8; ++v) {
    float inv = 1.0f / lrow[v];
    int qg = q0 + half * 8 + v;
    size_t base = ((size_t)b * SEQ + qg) * (size_t)DIM + (size_t)h * 64;
#pragma unroll
    for (int c = 0; c < 4; ++c)
      Out[base + c * 16 + l16] = (bf16)(o[c][v] * inv);
  }
}

// ---------------------------------------------------------------- launch
extern "C" void kernel_launch(void* const* d_in, const int* in_sizes, int n_in,
                              void* d_out, int out_size, void* d_ws, size_t ws_size,
                              hipStream_t stream) {
  const float* x  = (const float*)d_in[0];
  const float* wq = (const float*)d_in[1];
  const float* wk = (const float*)d_in[2];
  const float* wv = (const float*)d_in[3];
  const float* wo = (const float*)d_in[4];
  float* out = (float*)d_out;

  char* ws = (char*)d_ws;
  size_t off = 0;
  auto alloc = [&](size_t bytes) -> void* {
    void* p = ws + off;
    off += (bytes + 255) & ~(size_t)255;
    return p;
  };

  const size_t MS = (size_t)BATCH * SEQ;  // 4096 rows
  bf16* xb  = (bf16*)alloc(MS * DIM * 2);
  bf16* wqb = (bf16*)alloc((size_t)DIM * DIM * 2);
  bf16* wkb = (bf16*)alloc((size_t)DIM * 512 * 2);
  bf16* wvb = (bf16*)alloc((size_t)DIM * 512 * 2);
  bf16* wob = (bf16*)alloc((size_t)DIM * DIM * 2);
  float* qf = (float*)alloc(MS * DIM * 4);
  float* kf = (float*)alloc(MS * 512 * 4);
  float* vf = (float*)alloc(MS * 512 * 4);
  bf16* qb  = (bf16*)alloc(MS * DIM * 2);   // [B][32][S][64]
  bf16* kb  = (bf16*)alloc(MS * 512 * 2);   // [B][8][S][64]
  bf16* vb  = (bf16*)alloc(MS * 512 * 2);   // [B][8][S][64]
  bf16* ab  = (bf16*)qf;                    // alias: qf dead after RoPE

  // 1) fp32 -> bf16 conversions (vectorized x4)
  cvt_f32_bf16<<<2048, 256, 0, stream>>>(x,  xb,  (int)(MS * DIM / 4));
  cvt_f32_bf16<<<2048, 256, 0, stream>>>(wq, wqb, DIM * DIM / 4);
  cvt_f32_bf16<<<1024, 256, 0, stream>>>(wk, wkb, DIM * 512 / 4);
  cvt_f32_bf16<<<1024, 256, 0, stream>>>(wv, wvb, DIM * 512 / 4);
  cvt_f32_bf16<<<2048, 256, 0, stream>>>(wo, wob, DIM * DIM / 4);

  // 2) QKV projections (bf16 WMMA, f32 out)
  gemm_bf16<<<dim3(DIM / 128, MS / 128), 256, 0, stream>>>(xb, wqb, qf, (int)MS, DIM, DIM);
  gemm_bf16<<<dim3(512 / 128, MS / 128), 256, 0, stream>>>(xb, wkb, kf, (int)MS, 512, DIM);
  gemm_bf16<<<dim3(512 / 128, MS / 128), 256, 0, stream>>>(xb, wvb, vf, (int)MS, 512, DIM);

  // 3) RoPE + head-major relayout (bf16)
  rope_scatter<<<(BATCH * SEQ * NH * 32) / 256, 256, 0, stream>>>(qf, qb, NH);
  rope_scatter<<<(BATCH * SEQ * NKV * 32) / 256, 256, 0, stream>>>(kf, kb, NKV);
  v_scatter<<<(BATCH * SEQ * NKV * 64) / 256, 256, 0, stream>>>(vf, vb);

  // 4) causal flash attention (WMMA QK^T and P*V)
  flash_attn<<<dim3(SEQ / 128, NH, BATCH), 256, 0, stream>>>(qb, kb, vb, ab);

  // 5) output projection -> f32 d_out
  gemm_bf16<<<dim3(DIM / 128, MS / 128), 256, 0, stream>>>(ab, wob, out, (int)MS, DIM, DIM);
}
